// TemporalDecayAttention_21715354649845
// MI455X (gfx1250) — compile-verified, matrix operand order checked by reading, not run on hardware
//
#include <hip/hip_runtime.h>
#include <hip/hip_bf16.h>

#define D_EMBED 1024
#define N_HEADS 16
#define HD 64
#define LOG2_DECAY (-0.07400058144377693f)   // log2(0.95)

typedef __attribute__((ext_vector_type(16))) __bf16 bf16x16;
typedef __attribute__((ext_vector_type(8)))  __bf16 bf16x8;
typedef __attribute__((ext_vector_type(8)))  float  f32x8;

__device__ __forceinline__ f32x8 wmma_bf16(bf16x16 a, bf16x16 b, f32x8 c) {
  // D = A(16x32 bf16) * B(32x16 bf16) + C(16x16 f32)
  return __builtin_amdgcn_wmma_f32_16x16x32_bf16(
      /*neg_a=*/false, a, /*neg_b=*/false, b,
      /*c_mod=*/(short)0, c, /*reuse_a=*/false, /*reuse_b=*/false);
}

__device__ __forceinline__ void wait_asynccnt0() {
#if __has_builtin(__builtin_amdgcn_s_wait_asynccnt)
  __builtin_amdgcn_s_wait_asynccnt(0);
#else
  asm volatile("s_wait_asynccnt 0x0" ::: "memory");
#endif
}

// Async-copy 16B from global to LDS (tracked by ASYNCcnt, no VGPR round-trip).
__device__ __forceinline__ void async_ld_b128(const void* gptr, void* lds) {
  unsigned ldsoff = (unsigned)(size_t)lds;            // low 32 bits = LDS offset
  unsigned long long ga = (unsigned long long)gptr;
  asm volatile("global_load_async_to_lds_b128 %0, %1, off"
               :: "v"(ldsoff), "v"(ga) : "memory");
}

// Per-lane WMMA fragment load: elements are row[kbase+0..7] and
// row[kbase+16..23] (CDNA5 16-bit A/B layout; both chunks contiguous -> b128).
__device__ __forceinline__ bf16x16 frag_ld(const __bf16* row, int kbase) {
  bf16x16 f;
#pragma unroll
  for (int e = 0; e < 8; ++e) f[e] = row[kbase + e];
#pragma unroll
  for (int e = 0; e < 8; ++e) f[8 + e] = row[kbase + 16 + e];
  return f;
}

// Stage 16 K-contiguous elements from global into LDS (f32 -> bf16 downconvert).
__device__ __forceinline__ void stage16(const float* g, __bf16* s) {
  const float4* g4 = reinterpret_cast<const float4*>(g);
#pragma unroll
  for (int i = 0; i < 4; ++i) {
    float4 v = g4[i];
    s[4 * i + 0] = (__bf16)v.x; s[4 * i + 1] = (__bf16)v.y;
    s[4 * i + 2] = (__bf16)v.z; s[4 * i + 3] = (__bf16)v.w;
  }
}
__device__ __forceinline__ void stage16(const __bf16* g, __bf16* s) {
  const bf16x8* g8 = reinterpret_cast<const bf16x8*>(g);
#pragma unroll
  for (int i = 0; i < 2; ++i) {
    bf16x8 v = g8[i];
#pragma unroll
    for (int j = 0; j < 8; ++j) s[8 * i + j] = v[j];
  }
}

// ---------------------------------------------------------------------------
// C[M,N] = A[M,K] @ W[N,K]^T + bias[N]
// OUT_MODE 0: f32 row-major [M,N]
// OUT_MODE 1: bf16 head-split [B, H, S, HD]  (m = b*S+s, n = h*HD+hd)
// ---------------------------------------------------------------------------
template <int OUT_MODE, typename AT>
__global__ __launch_bounds__(256) void gemm_bt_kernel(
    const AT* __restrict__ A, const float* __restrict__ W,
    const float* __restrict__ bias, void* __restrict__ Cout,
    int M, int N, int K, int S) {
  constexpr int BM = 128, BN = 128, BK = 64, PITCH = 72;
  __shared__ __bf16 As[BM * PITCH];
  __shared__ __bf16 Bs[BN * PITCH];

  const int tid  = threadIdx.x;
  const int lane = tid & 31, wave = tid >> 5;
  const int wr = wave >> 1, wc = wave & 1;       // waves 4x2 over 128x128 tile
  const int m0 = blockIdx.y * BM, n0 = blockIdx.x * BN;
  const int rl = lane & 15, hi = lane >> 4;
  const int kb = 8 * hi;

  f32x8 acc[2][4];
#pragma unroll
  for (int mi = 0; mi < 2; ++mi)
#pragma unroll
    for (int ni = 0; ni < 4; ++ni)
#pragma unroll
      for (int r = 0; r < 8; ++r) acc[mi][ni][r] = 0.f;

  const int lr = tid >> 1;        // staging: row 0..127
  const int lc = (tid & 1) * 32;  // staging: col 0 or 32

  for (int kt = 0; kt < K; kt += BK) {
    const float* __restrict__ pfa = nullptr;  // silence template type for AT
    (void)pfa;
    stage16(A + (size_t)(m0 + lr) * K + kt + lc,      &As[lr * PITCH + lc]);
    stage16(A + (size_t)(m0 + lr) * K + kt + lc + 16, &As[lr * PITCH + lc + 16]);
    stage16(W + (size_t)(n0 + lr) * K + kt + lc,      &Bs[lr * PITCH + lc]);
    stage16(W + (size_t)(n0 + lr) * K + kt + lc + 16, &Bs[lr * PITCH + lc + 16]);
    if (kt + BK < K) {  // pull next tiles toward L2/L0 (global_prefetch_b8)
      __builtin_prefetch(A + (size_t)(m0 + lr) * K + kt + BK + lc, 0, 3);
      __builtin_prefetch(W + (size_t)(n0 + lr) * K + kt + BK + lc, 0, 3);
    }
    __syncthreads();

#pragma unroll
    for (int kk = 0; kk < 2; ++kk) {
      const int kb2 = kb + 32 * kk;
      bf16x16 af[2], bfr[4];
#pragma unroll
      for (int mi = 0; mi < 2; ++mi)
        af[mi] = frag_ld(&As[(wr * 32 + mi * 16 + rl) * PITCH], kb2);
#pragma unroll
      for (int ni = 0; ni < 4; ++ni)
        bfr[ni] = frag_ld(&Bs[(wc * 64 + ni * 16 + rl) * PITCH], kb2);
#pragma unroll
      for (int mi = 0; mi < 2; ++mi)
#pragma unroll
        for (int ni = 0; ni < 4; ++ni)
          acc[mi][ni] = wmma_bf16(af[mi], bfr[ni], acc[mi][ni]);
    }
    __syncthreads();
  }

#pragma unroll
  for (int mi = 0; mi < 2; ++mi)
#pragma unroll
    for (int ni = 0; ni < 4; ++ni)
#pragma unroll
      for (int r = 0; r < 8; ++r) {
        int m = m0 + wr * 32 + mi * 16 + r + 8 * hi;  // C layout: VGPR r -> M=r / r+8
        int n = n0 + wc * 64 + ni * 16 + rl;          // N = lane&15
        float val = acc[mi][ni][r] + bias[n];
        if (OUT_MODE == 0) {
          ((float*)Cout)[(size_t)m * N + n] = val;
        } else {
          int b = m / S, srow = m % S;
          int h = n >> 6, hd = n & 63;
          ((__bf16*)Cout)[((size_t)(b * N_HEADS + h) * S + srow) * HD + hd] =
              (__bf16)val;
        }
      }
}

// ---------------------------------------------------------------------------
// Flash-style attention with multiplicative temporal decay mask.
// Grid: (S/64, B*H); 128 threads = 4 waves, 16 q-rows per wave.
// K tile staged with GLOBAL_LOAD_ASYNC_TO_LDS_B128 (ASYNCcnt);
// V tile staged transposed through VGPRs (async can't transpose).
// ---------------------------------------------------------------------------
__global__ __launch_bounds__(128) void attn_decay_kernel(
    const __bf16* __restrict__ Q, const __bf16* __restrict__ Kx,
    const __bf16* __restrict__ V, __bf16* __restrict__ O, int S) {
  constexpr int KP = 72;  // LDS pitch (bank stagger, keeps 16B chunk alignment)
  __shared__ __bf16 Ks[64 * KP];       // [kv][hd]   (async-filled)
  __shared__ __bf16 VsT[64 * KP];      // [hd][kv]   (transposed on stage)
  __shared__ __bf16 Ps[4 * 16 * KP];   // per-wave P tile [16][64]

  const int tid = threadIdx.x, lane = tid & 31, wave = tid >> 5;
  const int rl = lane & 15, hi = lane >> 4;
  const int bh = blockIdx.y;
  const int q0 = blockIdx.x * 64 + wave * 16;
  const size_t base = (size_t)bh * S * HD;

  // Q fragments live in registers for the whole sweep (Hd=64 -> 2 frags).
  bf16x16 qf[2];
#pragma unroll
  for (int kk = 0; kk < 2; ++kk)
    qf[kk] = frag_ld(Q + base + (size_t)(q0 + rl) * HD, 8 * hi + 32 * kk);

  float mrun[8], lrun[8];
  f32x8 oacc[4];
#pragma unroll
  for (int r = 0; r < 8; ++r) { mrun[r] = -1e30f; lrun[r] = 0.f; }
#pragma unroll
  for (int ni = 0; ni < 4; ++ni)
#pragma unroll
    for (int r = 0; r < 8; ++r) oacc[ni][r] = 0.f;

  const int lr = tid >> 1;        // staging: kv row 0..63
  const int lc = (tid & 1) * 32;  // staging: hd col 0 or 32

  for (int kv0 = 0; kv0 < S; kv0 += 64) {
    __syncthreads();  // protect Ks/VsT from previous iteration's readers

    // K tile: 4x 16B async copies per thread, straight into LDS
    {
      const __bf16* gk = Kx + base + (size_t)(kv0 + lr) * HD;
#pragma unroll
      for (int c = 0; c < 4; ++c) {
        int col = lc + c * 8;
        async_ld_b128(gk + col, &Ks[lr * KP + col]);
      }
    }
    // V tile: load through VGPRs and store transposed
    {
      const bf16x8* gv = (const bf16x8*)(V + base + (size_t)(kv0 + lr) * HD + lc);
#pragma unroll
      for (int i = 0; i < 4; ++i) {
        bf16x8 vv8 = gv[i];
#pragma unroll
        for (int j = 0; j < 8; ++j)
          VsT[(lc + 8 * i + j) * KP + lr] = vv8[j];  // transpose
      }
    }
    wait_asynccnt0();
    __syncthreads();

    // S = Q @ K^T  (B-frag element B[k=hd][n=kv] = Ks[n][k]: contiguous in k)
    f32x8 sc[4];
#pragma unroll
    for (int ni = 0; ni < 4; ++ni) {
#pragma unroll
      for (int r = 0; r < 8; ++r) sc[ni][r] = 0.f;
#pragma unroll
      for (int kk = 0; kk < 2; ++kk) {
        bf16x16 kfrag = frag_ld(&Ks[(ni * 16 + rl) * KP], 8 * hi + 32 * kk);
        sc[ni] = wmma_bf16(qf[kk], kfrag, sc[ni]);
      }
    }

    // scale, multiplicative decay, online softmax
    float tmax[8];
#pragma unroll
    for (int r = 0; r < 8; ++r) tmax[r] = -1e30f;
#pragma unroll
    for (int ni = 0; ni < 4; ++ni)
#pragma unroll
      for (int r = 0; r < 8; ++r) {
        int qr = q0 + r + 8 * hi;
        int kc = kv0 + ni * 16 + rl;
        float dd = fabsf((float)(qr - kc));
        float sval = sc[ni][r] * 0.125f * exp2f(dd * LOG2_DECAY);
        sc[ni][r] = sval;
        tmax[r] = fmaxf(tmax[r], sval);
      }
#pragma unroll
    for (int r = 0; r < 8; ++r) {
#pragma unroll
      for (int off = 8; off >= 1; off >>= 1)  // reduce within 16-lane half
        tmax[r] = fmaxf(tmax[r], __shfl_xor(tmax[r], off, 32));
      float mnew = fmaxf(mrun[r], tmax[r]);
      float corr = __expf(mrun[r] - mnew);
      float ps = 0.f;
#pragma unroll
      for (int ni = 0; ni < 4; ++ni) {
        float p = __expf(sc[ni][r] - mnew);
        sc[ni][r] = p;
        ps += p;
      }
#pragma unroll
      for (int off = 8; off >= 1; off >>= 1)
        ps += __shfl_xor(ps, off, 32);
      lrun[r] = lrun[r] * corr + ps;
      mrun[r] = mnew;
#pragma unroll
      for (int ni = 0; ni < 4; ++ni) oacc[ni][r] *= corr;
    }

    // C-layout -> A-layout reshape of P via LDS
    __bf16* pw = &Ps[wave * 16 * KP];
#pragma unroll
    for (int ni = 0; ni < 4; ++ni)
#pragma unroll
      for (int r = 0; r < 8; ++r)
        pw[(r + 8 * hi) * KP + ni * 16 + rl] = (__bf16)sc[ni][r];
    __syncthreads();

    // O += P @ V  (B-frag element B[k=kv][n=hd] = VsT[n][k]: contiguous in k)
#pragma unroll
    for (int kk = 0; kk < 2; ++kk) {
      bf16x16 pf = frag_ld(&pw[rl * KP], 8 * hi + 32 * kk);
#pragma unroll
      for (int ni = 0; ni < 4; ++ni) {
        bf16x16 vfrag = frag_ld(&VsT[(ni * 16 + rl) * KP], 8 * hi + 32 * kk);
        oacc[ni] = wmma_bf16(pf, vfrag, oacc[ni]);
      }
    }
  }

  // normalize + write bf16 [B,S,D] (row-major for the final projection)
  const int b = bh / N_HEADS, h = bh % N_HEADS;
#pragma unroll
  for (int ni = 0; ni < 4; ++ni)
#pragma unroll
    for (int r = 0; r < 8; ++r) {
      int qr = q0 + r + 8 * hi;
      float val = oacc[ni][r] / lrun[r];
      O[((size_t)b * S + qr) * D_EMBED + h * HD + ni * 16 + rl] = (__bf16)val;
    }
}

// ---------------------------------------------------------------------------
extern "C" void kernel_launch(void* const* d_in, const int* in_sizes, int n_in,
                              void* d_out, int out_size, void* d_ws, size_t ws_size,
                              hipStream_t stream) {
  const float* x  = (const float*)d_in[0];
  const float* Wq = (const float*)d_in[1];
  const float* bq = (const float*)d_in[2];
  const float* Wk = (const float*)d_in[3];
  const float* bk = (const float*)d_in[4];
  const float* Wv = (const float*)d_in[5];
  const float* bv = (const float*)d_in[6];
  const float* Wo = (const float*)d_in[7];
  const float* bo = (const float*)d_in[8];

  const int B = 2;
  const int S = in_sizes[0] / (B * D_EMBED);  // 2048
  const int M = B * S;                        // 4096

  __bf16* q_ws    = (__bf16*)d_ws;                     // [B,H,S,HD] bf16
  __bf16* k_ws    = q_ws    + (size_t)M * D_EMBED;
  __bf16* v_ws    = k_ws    + (size_t)M * D_EMBED;
  __bf16* attn_ws = v_ws    + (size_t)M * D_EMBED;     // [B,S,D] bf16

  dim3 ggrid(D_EMBED / 128, M / 128);
  gemm_bt_kernel<1, float><<<ggrid, 256, 0, stream>>>(x, Wq, bq, q_ws, M, D_EMBED, D_EMBED, S);
  gemm_bt_kernel<1, float><<<ggrid, 256, 0, stream>>>(x, Wk, bk, k_ws, M, D_EMBED, D_EMBED, S);
  gemm_bt_kernel<1, float><<<ggrid, 256, 0, stream>>>(x, Wv, bv, v_ws, M, D_EMBED, D_EMBED, S);

  attn_decay_kernel<<<dim3(S / 64, B * N_HEADS), 128, 0, stream>>>(
      q_ws, k_ws, v_ws, attn_ws, S);

  gemm_bt_kernel<0, __bf16><<<ggrid, 256, 0, stream>>>(
      attn_ws, Wo, bo, d_out, M, D_EMBED, D_EMBED, S);
}
